// AttentionEncoder_26225070309641
// MI455X (gfx1250) — matrix operand order repaired and checked by reading, not compiled, not run on hardware
//
#include <hip/hip_runtime.h>

// ---------------------------------------------------------------------------
// AttentionEncoder forward for MI455X (gfx1250, wave32, WMMA).
// All GEMMs run through v_wmma_f32_16x16x32_bf16 (bf16 inputs, f32 accum).
// Weights are pre-transposed to [N][K] bf16 so each lane's WMMA B-fragment is
// one contiguous 32B load; A-fragments are two contiguous 16B loads per lane
// (contiguous k = 16g+s, same order in A and B). K/V projection + attention is fused
// per batch row through LDS so K/V never hit HBM. The batched GEMM uses a
// 32x64 wave tile with a ping-pong (unroll-2) software pipeline: no register
// rotation copies, loads of step k+32 issued before the 8 WMMAs of step k.
//
// d_in leaf order assumption: JAX tree_flatten (dict keys sorted alphabetically,
// {'b','w'} -> b,w ; {'b','g'} -> b,g ; block keys fc1,fc2,k,ln1,ln2,o,q,v).
//   0: obs
//   1..16 : nb_blocks[0]   17..32 : nb_blocks[1]
//   33,34 nb_cache b,w  35,36 nb_ln b,g  37,38 nb_pos b,w  39,40 nb_proc b,w
//   41,42 nb_query b,w  43,44 uav_active b,w  45,46 uav_cache b,w
//   47,48 uav_ln b,g  49,50 uav_pos b,w
//   51..66: ue_blocks[0]  67..82: ue_blocks[1]
//   83 ue_file_table  84,85 ue_hit b,w  86,87 ue_ln b,g  88,89 ue_pos b,w
//   90,91 ue_query b,w
// ---------------------------------------------------------------------------

#define B_SZ 4096
#define OBS  4774

typedef __attribute__((ext_vector_type(16))) __bf16        bf16x16;
typedef __attribute__((ext_vector_type(8)))  float         f32x8;
typedef __attribute__((ext_vector_type(4)))  unsigned int  u32x4;
typedef unsigned short u16;

static __device__ __forceinline__ u16 f2bf(float f) {
  unsigned u = __float_as_uint(f);
  u += 0x7FFFu + ((u >> 16) & 1u);     // round-to-nearest-even
  return (u16)(u >> 16);
}
static __device__ __forceinline__ float bf2f(u16 h) {
  return __uint_as_float(((unsigned)h) << 16);
}
static __device__ __forceinline__ float silu(float x) {
  return x / (1.f + __expf(-x));
}

union Frag { bf16x16 v; u32x4 q[2]; u16 s[16]; };

// A (16x32 bf16) fragment: lane L (row M = L%16, g = L/16) holds
// K = k0+16g..k0+16g+15 contiguously in elements 0..15.
static __device__ __forceinline__ bf16x16 load_a_frag(const u16* row, int g) {
  Frag f;
  f.q[0] = *(const u32x4*)(row + 16 * g);
  f.q[1] = *(const u32x4*)(row + 16 * g + 8);
  return f.v;
}
// B (32x16 bf16) fragment: lane L (col N = L%16, g = L/16) holds K = k0+16g..+15
// contiguously -> pass pointer already offset by n*K + k0 + 16g.
static __device__ __forceinline__ bf16x16 load_b_frag(const u16* colk) {
  Frag f;
  f.q[0] = *(const u32x4*)(colk);
  f.q[1] = *(const u32x4*)(colk + 8);
  return f.v;
}
static __device__ __forceinline__ f32x8 wmma_bf16(bf16x16 a, bf16x16 b, f32x8 c) {
  return __builtin_amdgcn_wmma_f32_16x16x32_bf16(false, a, false, b, (short)0, c,
                                                 false, false);
}

static __device__ __forceinline__ float blk_sum256(float v, float* sb) {
  const int t = threadIdx.x;
  sb[t] = v;
  __syncthreads();
#pragma unroll
  for (int s = 128; s > 0; s >>= 1) {
    if (t < s) sb[t] += sb[t + s];
    __syncthreads();
  }
  float r = sb[0];
  __syncthreads();
  return r;
}

// ---------------------------------------------------------------------------
// Weight prep: fp32 [K][N] row-major -> bf16 [N][K] ("B-friendly" layout)
// ---------------------------------------------------------------------------
__global__ void k_transpose(const float* __restrict__ w, u16* __restrict__ o,
                            int K, int N) {
  int idx = blockIdx.x * 256 + threadIdx.x;
  if (idx >= K * N) return;
  int k = idx / N, n = idx % N;
  o[(size_t)n * K + k] = f2bf(w[(size_t)k * N + n]);
}

// ---------------------------------------------------------------------------
// Extract bf16 GEMM inputs from obs + integer masks
// ---------------------------------------------------------------------------
__global__ __launch_bounds__(256) void k_extract(const float* __restrict__ obs,
                                                 u16* __restrict__ uavA,
                                                 u16* __restrict__ nbA,
                                                 int* __restrict__ ucnt,
                                                 int* __restrict__ ncnt) {
  const int b = blockIdx.x, t = threadIdx.x;
  const float* o = obs + (size_t)b * OBS;
  uavA[(size_t)b * 256 + t] = f2bf(o[3 + t]);
#pragma unroll
  for (int tk = 0; tk < 16; ++tk)
    nbA[((size_t)b * 16 + tk) * 256 + t] = f2bf(o[260 + tk * 262 + 3 + t]);
  if (t == 0) {
    ncnt[b] = (int)o[4452];
    ucnt[b] = (int)o[4773];
  }
}

// ---------------------------------------------------------------------------
// Generic WMMA GEMM: C[M][N] = act(A[M][K] @ W[K][N] + bias)
// A: bf16 row-major, WT: bf16 [N][K], M%32==0, N%64==0, K%64==0.
// blockDim=256 (8 waves); each wave computes one 32x64 tile.
// Ping-pong pipeline (unroll-2): frag set P holds step k, set Q holds k+32;
// loads for the next step are issued before the current step's 8 WMMAs, and
// no register-rotation copies are needed.
// ---------------------------------------------------------------------------
__global__ __launch_bounds__(256) void k_gemm(const u16* __restrict__ A,
                                              const u16* __restrict__ WT,
                                              const float* __restrict__ bias,
                                              float* __restrict__ Cf,
                                              u16* __restrict__ Cb,
                                              int M, int N, int K, int act) {
  const int wid  = blockIdx.x * 8 + (threadIdx.x >> 5);
  const int lane = threadIdx.x & 31;
  const int g = lane >> 4, ln = lane & 15;
  const int tm = M >> 5;                     // 32-row tiles
  const int m0 = (wid % tm) << 5;
  const int n0 = (wid / tm) << 6;

  f32x8 acc[2][4];
#pragma unroll
  for (int j = 0; j < 4; ++j) {
    float bv = bias[n0 + 16 * j + ln];
#pragma unroll
    for (int r = 0; r < 8; ++r) { acc[0][j][r] = bv; acc[1][j][r] = bv; }
  }

  const u16* arow0 = A + (size_t)(m0 + ln) * K;
  const u16* arow1 = A + (size_t)(m0 + 16 + ln) * K;
  const u16* wcol  = WT + (size_t)(n0 + ln) * K + 16 * g;   // j advances by 16*K

  bf16x16 a0P, a1P, bP[4];   // frag set P
  bf16x16 a0Q, a1Q, bQ[4];   // frag set Q

  auto loadP = [&](int k) {
    a0P = load_a_frag(arow0 + k, g);
    a1P = load_a_frag(arow1 + k, g);
#pragma unroll
    for (int j = 0; j < 4; ++j) bP[j] = load_b_frag(wcol + (size_t)(16 * j) * K + k);
  };
  auto loadQ = [&](int k) {
    a0Q = load_a_frag(arow0 + k, g);
    a1Q = load_a_frag(arow1 + k, g);
#pragma unroll
    for (int j = 0; j < 4; ++j) bQ[j] = load_b_frag(wcol + (size_t)(16 * j) * K + k);
  };
  auto fmaP = [&]() {
#pragma unroll
    for (int j = 0; j < 4; ++j) {
      acc[0][j] = wmma_bf16(a0P, bP[j], acc[0][j]);
      acc[1][j] = wmma_bf16(a1P, bP[j], acc[1][j]);
    }
  };
  auto fmaQ = [&]() {
#pragma unroll
    for (int j = 0; j < 4; ++j) {
      acc[0][j] = wmma_bf16(a0Q, bQ[j], acc[0][j]);
      acc[1][j] = wmma_bf16(a1Q, bQ[j], acc[1][j]);
    }
  };

  // K is a multiple of 64, so the k-steps pair up exactly.
  loadP(0);
  int k = 0;
  for (; k + 64 < K; k += 64) {
    loadQ(k + 32);   // issue loads, hidden behind fmaP's 8 WMMAs
    fmaP();
    loadP(k + 64);   // issue loads, hidden behind fmaQ's 8 WMMAs
    fmaQ();
  }
  loadQ(K - 32);     // k == K-64 here
  fmaP();
  fmaQ();

#pragma unroll
  for (int h = 0; h < 2; ++h) {
#pragma unroll
    for (int j = 0; j < 4; ++j) {
      const int col = n0 + 16 * j + ln;
#pragma unroll
      for (int r = 0; r < 8; ++r) {
        const int row = m0 + 16 * h + r + 8 * g;  // C/D: VGPR r -> M=r+8g, N=ln
        float v = acc[h][j][r];
        if (act) v = silu(v);
        size_t o = (size_t)row * N + col;
        if (Cf) Cf[o] = v;
        if (Cb) Cb[o] = f2bf(v);
      }
    }
  }
}

// ---------------------------------------------------------------------------
// uav embedding: concat(silu(pos@W),silu(cacheproj),silu(act@W)) -> LN
// ---------------------------------------------------------------------------
__global__ __launch_bounds__(256) void k_uav_emb(
    const float* __restrict__ obs, const float* __restrict__ wpos,
    const float* __restrict__ bpos, const float* __restrict__ cp,
    const float* __restrict__ wact, const float* __restrict__ bact,
    const float* __restrict__ g, const float* __restrict__ be,
    float* __restrict__ embF, u16* __restrict__ embB) {
  __shared__ float sb[256];
  const int b = blockIdx.x, t = threadIdx.x;
  const float* o = obs + (size_t)b * OBS;
  float x;
  if (t < 64) {
    x = silu(o[0] * wpos[t] + o[1] * wpos[64 + t] + o[2] * wpos[128 + t] + bpos[t]);
  } else if (t < 192) {
    x = cp[(size_t)b * 128 + (t - 64)];      // already silu'd by GEMM
  } else {
    int j = t - 192;
    x = silu(o[259] * wact[j] + bact[j]);
  }
  float m = blk_sum256(x, sb) * (1.f / 256.f);
  float d = x - m;
  float v = blk_sum256(d * d, sb) * (1.f / 256.f);
  float y = d * rsqrtf(v + 1e-5f) * g[t] + be[t];
  embF[(size_t)b * 256 + t] = y;
  embB[(size_t)b * 256 + t] = f2bf(y);
}

// ---------------------------------------------------------------------------
// UE token embedding (one block per (b,s) token)
// ---------------------------------------------------------------------------
__global__ __launch_bounds__(256) void k_ue_tok(
    const float* __restrict__ obs, const float* __restrict__ wpos,
    const float* __restrict__ bpos, const float* __restrict__ ftab,
    const float* __restrict__ whit, const float* __restrict__ bhit,
    const float* __restrict__ g, const float* __restrict__ be,
    u16* __restrict__ tok) {
  __shared__ float sb[256];
  const int id = blockIdx.x;
  const int b = id >> 6, s = id & 63;
  const int t = threadIdx.x;
  const float* f = obs + (size_t)b * OBS + 4453 + s * 5;
  float x;
  if (t < 128) {
    x = silu(f[0] * wpos[t] + f[1] * wpos[128 + t] + f[2] * wpos[256 + t] + bpos[t]);
  } else if (t < 192) {
    int fid = (int)rintf(f[3] * 256.f);
    fid = fid < 0 ? 0 : (fid > 255 ? 255 : fid);
    x = ftab[(size_t)fid * 64 + (t - 128)];
  } else {
    int j = t - 192;
    x = silu(f[4] * whit[j] + bhit[j]);
  }
  float m = blk_sum256(x, sb) * (1.f / 256.f);
  float d = x - m;
  float v = blk_sum256(d * d, sb) * (1.f / 256.f);
  float y = d * rsqrtf(v + 1e-5f) * g[t] + be[t];
  tok[(size_t)id * 256 + t] = f2bf(y);
}

// ---------------------------------------------------------------------------
// NB token embedding (one block per (b,tk) token)
// ---------------------------------------------------------------------------
__global__ __launch_bounds__(256) void k_nb_tok(
    const float* __restrict__ obs, const float* __restrict__ wpos,
    const float* __restrict__ bpos, const float* __restrict__ cp,
    const float* __restrict__ wproc, const float* __restrict__ bproc,
    const float* __restrict__ g, const float* __restrict__ be,
    u16* __restrict__ tok) {
  __shared__ float sb[256];
  const int id = blockIdx.x;
  const int b = id >> 4, tk = id & 15;
  const int t = threadIdx.x;
  const float* f = obs + (size_t)b * OBS + 260 + tk * 262;
  float x;
  if (t < 64) {
    x = silu(f[0] * wpos[t] + f[1] * wpos[64 + t] + f[2] * wpos[128 + t] + bpos[t]);
  } else if (t < 192) {
    x = cp[(size_t)id * 128 + (t - 64)];     // already silu'd by GEMM
  } else {
    int j = t - 192;
    x = silu(f[259] * wproc[j] + f[260] * wproc[64 + j] + f[261] * wproc[128 + j] +
             bproc[j]);
  }
  float m = blk_sum256(x, sb) * (1.f / 256.f);
  float d = x - m;
  float v = blk_sum256(d * d, sb) * (1.f / 256.f);
  float y = d * rsqrtf(v + 1e-5f) * g[t] + be[t];
  tok[(size_t)id * 256 + t] = f2bf(y);
}

// ---------------------------------------------------------------------------
// Fused K/V projection + masked multi-head attention (per batch row).
// Phase1: K = tok@Wk+bk (WMMA) -> LDS.  Phase2: scores/softmax (masked).
// Phase3: V = tok@Wv+bv (WMMA) -> LDS (reusing K's space).  Phase4: out = w·V.
// ---------------------------------------------------------------------------
template <int S>
__global__ __launch_bounds__(256) void k_kv_attn(
    const u16* __restrict__ tok, const u16* __restrict__ WkT,
    const float* __restrict__ bk, const u16* __restrict__ WvT,
    const float* __restrict__ bv, const u16* __restrict__ Q,
    const int* __restrict__ cnt, u16* __restrict__ attn) {
  __shared__ u16 sM[S * 256];
  __shared__ float sQ[256];
  __shared__ float ssc[4 * S];
  __shared__ float sred[8];

  const int b = blockIdx.x;
  const int t = threadIdx.x;
  const int wave = t >> 5, lane = t & 31;
  const int g = lane >> 4, ln = lane & 15;
  constexpr int TM = S / 16;
  constexpr int TPW = (TM * 16) / 8;   // tiles per wave

  sQ[t] = bf2f(Q[(size_t)b * 256 + t]);
  const u16* tb = tok + (size_t)b * S * 256;

  // ---- phase 1: K tiles -> LDS ----
  for (int i = 0; i < TPW; ++i) {
    const int tt = wave * TPW + i;
    const int m0 = (tt % TM) * 16, n0 = (tt / TM) * 16;
    f32x8 acc;
    float bvv = bk[n0 + ln];
#pragma unroll
    for (int r = 0; r < 8; ++r) acc[r] = bvv;
    const u16* arow = tb + (size_t)(m0 + ln) * 256;
#pragma unroll
    for (int k0 = 0; k0 < 256; k0 += 32) {
      bf16x16 a = load_a_frag(arow + k0, g);
      bf16x16 w = load_b_frag(WkT + (size_t)(n0 + ln) * 256 + k0 + 16 * g);
      acc = wmma_bf16(a, w, acc);
    }
#pragma unroll
    for (int r = 0; r < 8; ++r) sM[(m0 + r + 8 * g) * 256 + n0 + ln] = f2bf(acc[r]);
  }
  __syncthreads();

  // ---- phase 2: masked scores + softmax ----
  const int c = cnt[b];
  if (t < 4 * S) {
    const int h = t / S, s = t % S;
    float sc = -__builtin_inff();
    if (s < c) {
      float a0 = 0.f;
      const u16* kp = &sM[s * 256 + h * 64];
#pragma unroll 8
      for (int d = 0; d < 64; ++d) a0 += sQ[h * 64 + d] * bf2f(kp[d]);
      sc = a0 * 0.125f;                 // 1/sqrt(64)
    }
    ssc[t] = sc;
  }
  __syncthreads();
  if (t < 4) {
    float m = -__builtin_inff();
    for (int s = 0; s < S; ++s) m = fmaxf(m, ssc[t * S + s]);
    sred[t] = m;
  }
  __syncthreads();
  if (t < 4 * S) {
    const int h = t / S;
    float scv = ssc[t];
    ssc[t] = (scv > -__builtin_inff()) ? __expf(scv - sred[h]) : 0.f;
  }
  __syncthreads();
  if (t < 4) {
    float sm = 0.f;
    for (int s = 0; s < S; ++s) sm += ssc[t * S + s];
    sred[4 + t] = sm;
  }
  __syncthreads();
  if (t < 4 * S) {
    const int h = t / S;
    float sm = sred[4 + h];
    ssc[t] = (sm > 0.f) ? ssc[t] / sm : 0.f;   // cnt==0 -> all-zero weights
  }
  __syncthreads();

  // ---- phase 3: V tiles -> LDS (reuse sM) ----
  for (int i = 0; i < TPW; ++i) {
    const int tt = wave * TPW + i;
    const int m0 = (tt % TM) * 16, n0 = (tt / TM) * 16;
    f32x8 acc;
    float bvv = bv[n0 + ln];
#pragma unroll
    for (int r = 0; r < 8; ++r) acc[r] = bvv;
    const u16* arow = tb + (size_t)(m0 + ln) * 256;
#pragma unroll
    for (int k0 = 0; k0 < 256; k0 += 32) {
      bf16x16 a = load_a_frag(arow + k0, g);
      bf16x16 w = load_b_frag(WvT + (size_t)(n0 + ln) * 256 + k0 + 16 * g);
      acc = wmma_bf16(a, w, acc);
    }
#pragma unroll
    for (int r = 0; r < 8; ++r) sM[(m0 + r + 8 * g) * 256 + n0 + ln] = f2bf(acc[r]);
  }
  __syncthreads();

  // ---- phase 4: weighted sum over V ----
  {
    const int h = t >> 6, d = t & 63;
    float o = 0.f;
#pragma unroll 4
    for (int s = 0; s < S; ++s) o += ssc[h * S + s] * bf2f(sM[s * 256 + h * 64 + d]);
    attn[(size_t)b * 256 + t] = f2bf(o);
  }
}

// ---------------------------------------------------------------------------
// x = LayerNorm(a + r) * g + be ; dual f32/bf16 output
// ---------------------------------------------------------------------------
__global__ __launch_bounds__(256) void k_ln_res(
    const float* __restrict__ a, const float* __restrict__ r,
    const float* __restrict__ g, const float* __restrict__ be,
    float* __restrict__ of, u16* __restrict__ ob) {
  __shared__ float sb[256];
  const int b = blockIdx.x, t = threadIdx.x;
  float x = a[(size_t)b * 256 + t] + r[(size_t)b * 256 + t];
  float m = blk_sum256(x, sb) * (1.f / 256.f);
  float d = x - m;
  float v = blk_sum256(d * d, sb) * (1.f / 256.f);
  float y = d * rsqrtf(v + 1e-5f) * g[t] + be[t];
  of[(size_t)b * 256 + t] = y;
  ob[(size_t)b * 256 + t] = f2bf(y);
}

// ---------------------------------------------------------------------------
// out = concat(uav_emb, q_ue * has_ue, q_nb * has_nb)
// ---------------------------------------------------------------------------
__global__ __launch_bounds__(256) void k_final(
    const float* __restrict__ emb, const float* __restrict__ qu,
    const float* __restrict__ qn, const int* __restrict__ uc,
    const int* __restrict__ nc, float* __restrict__ out) {
  const int b = blockIdx.x, t = threadIdx.x;
  const float mu = (uc[b] > 0) ? 1.f : 0.f;
  const float mn = (nc[b] > 0) ? 1.f : 0.f;
  out[(size_t)b * 768 + t]       = emb[(size_t)b * 256 + t];
  out[(size_t)b * 768 + 256 + t] = qu[(size_t)b * 256 + t] * mu;
  out[(size_t)b * 768 + 512 + t] = qn[(size_t)b * 256 + t] * mn;
}

// ---------------------------------------------------------------------------
extern "C" void kernel_launch(void* const* d_in, const int* in_sizes, int n_in,
                              void* d_out, int out_size, void* d_ws, size_t ws_size,
                              hipStream_t stream) {
  (void)in_sizes; (void)n_in; (void)out_size;
  const float* obs = (const float*)d_in[0];
  auto F = [&](int i) -> const float* { return (const float*)d_in[i]; };

  const int NBB[2] = {1, 17};
  const int UEB[2] = {51, 67};
  enum { FC1B = 0, FC1W, FC2B, FC2W, KB, KW, LN1B, LN1G, LN2B, LN2G,
         OB, OW, QB, QW, VB, VW };
  const int I_NB_CACHE_B = 33, I_NB_CACHE_W = 34, I_NB_LN_B = 35, I_NB_LN_G = 36,
            I_NB_POS_B = 37, I_NB_POS_W = 38, I_NB_PROC_B = 39, I_NB_PROC_W = 40,
            I_NB_QUERY_B = 41, I_NB_QUERY_W = 42, I_UAV_ACT_B = 43,
            I_UAV_ACT_W = 44, I_UAV_CACHE_B = 45, I_UAV_CACHE_W = 46,
            I_UAV_LN_B = 47, I_UAV_LN_G = 48, I_UAV_POS_B = 49, I_UAV_POS_W = 50,
            I_UE_FTAB = 83, I_UE_HIT_B = 84, I_UE_HIT_W = 85, I_UE_LN_B = 86,
            I_UE_LN_G = 87, I_UE_POS_B = 88, I_UE_POS_W = 89, I_UE_QUERY_B = 90,
            I_UE_QUERY_W = 91;

  char* ws = (char*)d_ws;
  size_t off = 0;
  auto alloc = [&](size_t bytes) -> void* {
    void* p = ws + off;
    off = (off + bytes + 255) & ~(size_t)255;
    return p;
  };

  u16* wtT[2][2][6];  // [path 0=ue,1=nb][blk][q,k,v,o,fc1,fc2]
  for (int p = 0; p < 2; ++p)
    for (int b = 0; b < 2; ++b) {
      for (int j = 0; j < 4; ++j) wtT[p][b][j] = (u16*)alloc(256 * 256 * 2);
      wtT[p][b][4] = (u16*)alloc(1024 * 256 * 2);
      wtT[p][b][5] = (u16*)alloc(256 * 1024 * 2);
    }
  u16* queryT[2];
  queryT[0] = (u16*)alloc(256 * 256 * 2);
  queryT[1] = (u16*)alloc(256 * 256 * 2);
  u16* cacheTU = (u16*)alloc(128 * 256 * 2);
  u16* cacheTN = (u16*)alloc(128 * 256 * 2);

  u16*   uavA = (u16*)alloc((size_t)B_SZ * 256 * 2);
  u16*   nbA  = (u16*)alloc((size_t)B_SZ * 16 * 256 * 2);
  float* cpU  = (float*)alloc((size_t)B_SZ * 128 * 4);
  float* cpN  = (float*)alloc((size_t)B_SZ * 16 * 128 * 4);
  float* embF = (float*)alloc((size_t)B_SZ * 256 * 4);
  u16*   embB = (u16*)alloc((size_t)B_SZ * 256 * 2);
  u16*   ueTok = (u16*)alloc((size_t)B_SZ * 64 * 256 * 2);
  u16*   nbTok = (u16*)alloc((size_t)B_SZ * 16 * 256 * 2);
  float* qF[2];
  u16*   qB16[2];
  qF[0] = (float*)alloc((size_t)B_SZ * 256 * 4);
  qB16[0] = (u16*)alloc((size_t)B_SZ * 256 * 2);
  qF[1] = (float*)alloc((size_t)B_SZ * 256 * 4);
  qB16[1] = (u16*)alloc((size_t)B_SZ * 256 * 2);
  u16*   Qp    = (u16*)alloc((size_t)B_SZ * 256 * 2);
  u16*   attnB = (u16*)alloc((size_t)B_SZ * 256 * 2);
  float* Obuf  = (float*)alloc((size_t)B_SZ * 256 * 4);
  float* xF    = (float*)alloc((size_t)B_SZ * 256 * 4);
  u16*   xB    = (u16*)alloc((size_t)B_SZ * 256 * 2);
  u16*   hid   = (u16*)alloc((size_t)B_SZ * 1024 * 2);
  float* Ff    = (float*)alloc((size_t)B_SZ * 256 * 4);
  int*   ucnt  = (int*)alloc(B_SZ * 4);
  int*   ncnt  = (int*)alloc(B_SZ * 4);
  if (off > ws_size) return;  // workspace too small — bail deterministically

  auto tr = [&](const float* w, u16* dst, int K, int N) {
    int total = K * N;
    k_transpose<<<(total + 255) / 256, 256, 0, stream>>>(w, dst, K, N);
  };
  for (int p = 0; p < 2; ++p)
    for (int b = 0; b < 2; ++b) {
      int base = (p == 0 ? UEB[b] : NBB[b]);
      tr(F(base + QW), wtT[p][b][0], 256, 256);
      tr(F(base + KW), wtT[p][b][1], 256, 256);
      tr(F(base + VW), wtT[p][b][2], 256, 256);
      tr(F(base + OW), wtT[p][b][3], 256, 256);
      tr(F(base + FC1W), wtT[p][b][4], 256, 1024);
      tr(F(base + FC2W), wtT[p][b][5], 1024, 256);
    }
  tr(F(I_UE_QUERY_W), queryT[0], 256, 256);
  tr(F(I_NB_QUERY_W), queryT[1], 256, 256);
  tr(F(I_UAV_CACHE_W), cacheTU, 256, 128);
  tr(F(I_NB_CACHE_W), cacheTN, 256, 128);

  k_extract<<<B_SZ, 256, 0, stream>>>(obs, uavA, nbA, ucnt, ncnt);

  auto gemm = [&](const u16* A, const u16* WT, const float* bias, float* Cf,
                  u16* Cb, int M, int N, int K, int act) {
    int waves = (M / 32) * (N / 64);
    k_gemm<<<waves / 8, 256, 0, stream>>>(A, WT, bias, Cf, Cb, M, N, K, act);
  };

  gemm(uavA, cacheTU, F(I_UAV_CACHE_B), cpU, nullptr, B_SZ, 128, 256, 1);
  gemm(nbA, cacheTN, F(I_NB_CACHE_B), cpN, nullptr, B_SZ * 16, 128, 256, 1);

  k_uav_emb<<<B_SZ, 256, 0, stream>>>(obs, F(I_UAV_POS_W), F(I_UAV_POS_B), cpU,
                                      F(I_UAV_ACT_W), F(I_UAV_ACT_B),
                                      F(I_UAV_LN_G), F(I_UAV_LN_B), embF, embB);
  k_ue_tok<<<B_SZ * 64, 256, 0, stream>>>(obs, F(I_UE_POS_W), F(I_UE_POS_B),
                                          F(I_UE_FTAB), F(I_UE_HIT_W),
                                          F(I_UE_HIT_B), F(I_UE_LN_G),
                                          F(I_UE_LN_B), ueTok);
  k_nb_tok<<<B_SZ * 16, 256, 0, stream>>>(obs, F(I_NB_POS_W), F(I_NB_POS_B), cpN,
                                          F(I_NB_PROC_W), F(I_NB_PROC_B),
                                          F(I_NB_LN_G), F(I_NB_LN_B), nbTok);

  gemm(embB, queryT[0], F(I_UE_QUERY_B), qF[0], qB16[0], B_SZ, 256, 256, 0);
  gemm(embB, queryT[1], F(I_NB_QUERY_B), qF[1], qB16[1], B_SZ, 256, 256, 0);

  for (int p = 0; p < 2; ++p) {
    const u16* tok = (p == 0) ? ueTok : nbTok;
    const int* cnt = (p == 0) ? ucnt : ncnt;
    for (int b = 0; b < 2; ++b) {
      int base = (p == 0 ? UEB[b] : NBB[b]);
      gemm(qB16[p], wtT[p][b][0], F(base + QB), nullptr, Qp, B_SZ, 256, 256, 0);
      if (p == 0)
        k_kv_attn<64><<<B_SZ, 256, 0, stream>>>(tok, wtT[p][b][1], F(base + KB),
                                                wtT[p][b][2], F(base + VB), Qp,
                                                cnt, attnB);
      else
        k_kv_attn<16><<<B_SZ, 256, 0, stream>>>(tok, wtT[p][b][1], F(base + KB),
                                                wtT[p][b][2], F(base + VB), Qp,
                                                cnt, attnB);
      gemm(attnB, wtT[p][b][3], F(base + OB), Obuf, nullptr, B_SZ, 256, 256, 0);
      k_ln_res<<<B_SZ, 256, 0, stream>>>(Obuf, qF[p], F(base + LN1G),
                                         F(base + LN1B), xF, xB);
      gemm(xB, wtT[p][b][4], F(base + FC1B), nullptr, hid, B_SZ, 1024, 256, 1);
      gemm(hid, wtT[p][b][5], F(base + FC2B), Ff, nullptr, B_SZ, 256, 1024, 0);
      k_ln_res<<<B_SZ, 256, 0, stream>>>(xF, Ff, F(base + LN2G), F(base + LN2B),
                                         qF[p], qB16[p]);
    }
  }

  k_final<<<B_SZ, 256, 0, stream>>>(embF, qF[0], qF[1], ucnt, ncnt,
                                    (float*)d_out);
}
